// Decoder_41343355191486
// MI455X (gfx1250) — compile-verified, matrix operand order checked by reading, not run on hardware
//
#include <hip/hip_runtime.h>
#include <hip/hip_bf16.h>
#include <cstdint>
#include <cstddef>

// ---------------------------------------------------------------------------
// Tacotron2-style decoder for MI455X (gfx1250, wave32, WMMA).
// LSTM GEMMs use v_wmma_f32_16x16x32_bf16 with weights pre-packed into the
// ISA fragment layout (bf16, L2-resident across the 600-step scan).
// Round 3: K-loop split into three branch-free region segments, each
// software-pipelined -> inner loop is loads + prefetch + wmma only.
// ---------------------------------------------------------------------------

typedef __bf16 bf16;
typedef bf16  v16bf __attribute__((ext_vector_type(16)));
typedef float v8f   __attribute__((ext_vector_type(8)));

struct U32x8 { uint4 a, b; };   // 32 bytes == one v16bf fragment

#define B_    64
#define S_    512
#define D_    512
#define T_    600
#define MEL_  80
#define PRE_  256
#define HID_  1024
#define NG_   4096
#define ATT_  128
#define LOCF_ 32
#define LOCK_ 31
#define K1_   1792   // 256 (prenet) + 512 (ctx) + 1024 (h1)
#define K2_   2560   // 1024 (h1) + 512 (ctx) + 1024 (h2)

// ------------------------------ utility kernels ----------------------------

__global__ void zero_kernel(float* p, size_t n) {
  for (size_t i = blockIdx.x * (size_t)blockDim.x + threadIdx.x; i < n;
       i += (size_t)gridDim.x * blockDim.x)
    p[i] = 0.f;
}

__global__ void bias_sum_kernel(const float* __restrict__ a,
                                const float* __restrict__ b,
                                float* __restrict__ o, int n) {
  int i = blockIdx.x * blockDim.x + threadIdx.x;
  if (i < n) o[i] = a[i] + b[i];
}

// Pack fused [W_ih | W_hh] (row-major f32, N=4096) into bf16 WMMA B-fragment
// layout: frag(ntile,ktile) stores 16 contiguous bf16 per lane so the GEMM
// inner loop is two global_load_b128 per fragment.
__global__ void pack_w_kernel(const float* __restrict__ wih,
                              const float* __restrict__ whh,
                              bf16* __restrict__ out, int Kih, int Khh) {
  const int Ktot = Kih + Khh;
  const int nK = Ktot >> 5;
  const size_t total = (size_t)NG_ * Ktot;
  for (size_t p = blockIdx.x * (size_t)blockDim.x + threadIdx.x; p < total;
       p += (size_t)gridDim.x * blockDim.x) {
    int e     = (int)(p & 15);
    int lane  = (int)((p >> 4) & 31);
    size_t r  = p >> 9;
    int kt    = (int)(r % nK);
    int ntile = (int)(r / nK);
    int n     = ntile * 16 + (lane & 15);
    int base  = (lane & 16) ? 8 : 0;                 // ISA 16-bit B layout
    int kl    = (e < 8) ? (base + e) : (base + 16 + (e - 8));
    int k     = kt * 32 + kl;
    float v = (k < Kih) ? wih[(size_t)n * Kih + k]
                        : whh[(size_t)n * Khh + (k - Kih)];
    out[p] = (bf16)v;
  }
}

// PreNet over all frames (fully parallel in t): relu(relu(x@W1^T+b1)@W2^T+b2)
__global__ __launch_bounds__(256) void prenet_kernel(
    const float* __restrict__ inputs, const float* __restrict__ pw1,
    const float* __restrict__ pb1, const float* __restrict__ pw2,
    const float* __restrict__ pb2, bf16* __restrict__ pre16) {
  __shared__ float sx[MEL_];
  __shared__ float s1[PRE_];
  const int blk = blockIdx.x;          // t*B + b
  const int t = blk / B_, b = blk % B_;
  const int tid = threadIdx.x;
  if (tid < MEL_)
    sx[tid] = (t == 0) ? 0.f : inputs[((size_t)b * T_ + (t - 1)) * MEL_ + tid];
  __syncthreads();
  float a = pb1[tid];
  for (int k = 0; k < MEL_; ++k) a += pw1[tid * MEL_ + k] * sx[k];
  s1[tid] = fmaxf(a, 0.f);
  __syncthreads();
  float a2 = pb2[tid];
  for (int k = 0; k < PRE_; ++k) a2 += pw2[tid * PRE_ + k] * s1[k];
  pre16[((size_t)t * B_ + b) * PRE_ + tid] = (bf16)fmaxf(a2, 0.f);
}

// processed memory: pm[b,s,a] = sum_d enc[b,s,d] * wv[a,d]
__global__ void procmem_kernel(const float* __restrict__ enc,
                               const float* __restrict__ wv,
                               float* __restrict__ pm) {
  const size_t total = (size_t)B_ * S_ * ATT_;
  for (size_t i = blockIdx.x * (size_t)blockDim.x + threadIdx.x; i < total;
       i += (size_t)gridDim.x * blockDim.x) {
    int a = (int)(i % ATT_);
    size_t bs = i / ATT_;
    const float* er = enc + bs * D_;
    const float* wr = wv + (size_t)a * D_;
    float acc = 0.f;
    for (int k = 0; k < D_; ++k) acc += er[k] * wr[k];
    pm[i] = acc;
  }
}

// ------------------------- fused WMMA LSTM cell ----------------------------
// gates = A @ Wp^T + bsum ; A rows gathered from 3 bf16 regions
// (x|ctx|h), then the cell nonlinearity is applied in-kernel via LDS.
// grid: 64 blocks (16 gate columns each), block: 512 threads = 16 waves,
// wave (mt, gate) owns one 16x16 C tile.
// Region boundaries are multiples of 32 -> the K loop is split into three
// branch-free segments, each software-pipelined one tile ahead; the packed
// weight stream advances linearly across segments.
__global__ __launch_bounds__(512) void lstm_kernel(
    const bf16* __restrict__ Wp, const float* __restrict__ bsum,
    const bf16* __restrict__ r0, int len0, int st0,
    const bf16* __restrict__ r1, int len1, int st1,
    const bf16* __restrict__ r2, int len2, int st2,
    int Ktot,
    float* __restrict__ cstate, float* __restrict__ hout,
    bf16* __restrict__ hb16) {
  __shared__ float gsm[4][B_][16];
  const int tid  = threadIdx.x;
  const int wave = tid >> 5, lane = tid & 31;
  const int mt = wave & 3, gate = wave >> 2;
  const int m0 = mt << 4;
  const int nK = Ktot >> 5;
  const int ntile = gate * 64 + blockIdx.x;
  const int lm = lane & 15;
  const int hi = lane >> 4;
  const int hoff = hi * 8;        // ISA 16-bit A layout: lane>=16 -> K+8
  const int row = m0 + lm;        // A: M = lane%16 for both lane halves

  // per-region per-lane A pointers (advance 32 bf16 = 64B per K tile)
  const bf16* aseg[3];
  int segn[3];
  aseg[0] = r0 + (size_t)row * st0 + hoff;  segn[0] = len0 >> 5;
  aseg[1] = r1 + (size_t)row * st1 + hoff;  segn[1] = len1 >> 5;
  aseg[2] = r2 + (size_t)row * st2 + hoff;  segn[2] = len2 >> 5;

  // weight stream: 32 lanes * 16 bf16 = 1KB per tile, linear across segments
  const uint4* bq = (const uint4*)(Wp + ((size_t)ntile * nK) * 512 +
                                   (size_t)lane * 16);
  v8f acc = {};
#pragma unroll
  for (int sg = 0; sg < 3; ++sg) {
    const bf16* ap = aseg[sg];
    const int n = segn[sg];
    U32x8 Ac, Bc;
    Ac.a = *(const uint4*)ap;           // K = base+hoff .. +7
    Ac.b = *(const uint4*)(ap + 16);    // K = base+hoff+16 .. +23
    Bc.a = bq[0];
    Bc.b = bq[1];
    for (int i = 0; i < n - 1; ++i) {
      ap += 32;
      bq += 64;
      U32x8 An, Bn;
      An.a = *(const uint4*)ap;
      An.b = *(const uint4*)(ap + 16);
      Bn.a = bq[0];
      Bn.b = bq[1];
      __builtin_prefetch((const void*)(bq + 128), 0, 1);  // 2 tiles ahead
      acc = __builtin_amdgcn_wmma_f32_16x16x32_bf16(
          false, __builtin_bit_cast(v16bf, Ac), false,
          __builtin_bit_cast(v16bf, Bc), (short)0, acc, false, false);
      Ac = An;
      Bc = Bn;
    }
    acc = __builtin_amdgcn_wmma_f32_16x16x32_bf16(
        false, __builtin_bit_cast(v16bf, Ac), false,
        __builtin_bit_cast(v16bf, Bc), (short)0, acc, false, false);
    bq += 64;
  }

  // C layout: VGPR r -> M = m0 + r + 8*hi, N = lane%16
#pragma unroll
  for (int r = 0; r < 8; ++r) gsm[gate][m0 + r + hi * 8][lm] = acc[r];
  __syncthreads();
  const int jb = blockIdx.x << 4;
  for (int idx = tid; idx < B_ * 16; idx += 512) {
    int m = idx >> 4, col = idx & 15;
    int j = jb + col;
    float gi = gsm[0][m][col] + bsum[j];
    float gf = gsm[1][m][col] + bsum[HID_ + j];
    float gg = gsm[2][m][col] + bsum[2 * HID_ + j];
    float go = gsm[3][m][col] + bsum[3 * HID_ + j];
    float si = 1.f / (1.f + expf(-gi));
    float sf = 1.f / (1.f + expf(-gf));
    float so = 1.f / (1.f + expf(-go));
    float cn = sf * cstate[(size_t)m * HID_ + j] + si * tanhf(gg);
    float h  = so * tanhf(cn);
    cstate[(size_t)m * HID_ + j] = cn;
    hout[(size_t)m * HID_ + j]   = h;
    hb16[(size_t)m * HID_ + j]   = (bf16)h;    // staged for next GEMM
  }
}

// ---------------------- location-sensitive attention -----------------------
// one block per batch row; 512 threads = one per encoder position
__global__ __launch_bounds__(512) void attn_kernel(
    const float* __restrict__ h1, const float* __restrict__ wq,
    const float* __restrict__ lk, const float* __restrict__ wl,
    const float* __restrict__ vat, const float* __restrict__ pm,
    const float* __restrict__ enc,
    float* __restrict__ al, float* __restrict__ alc,
    float* __restrict__ ctx, bf16* __restrict__ ctxb,
    float* __restrict__ alout) {
  __shared__ float slk[LOCF_ * 2 * LOCK_];
  __shared__ float swl[ATT_ * LOCF_];
  __shared__ float sq[ATT_];
  __shared__ float sv[ATT_];
  __shared__ float sal[S_];
  __shared__ float sac[S_];
  __shared__ float red[512];
  const int b = blockIdx.x, tid = threadIdx.x;

  for (int i = tid; i < LOCF_ * 2 * LOCK_; i += 512) slk[i] = lk[i];
  for (int i = tid; i < ATT_ * LOCF_; i += 512) swl[i] = wl[i];
  if (tid < ATT_) sv[tid] = vat[tid];
  sal[tid] = al[b * S_ + tid];
  sac[tid] = alc[b * S_ + tid];

  // q = wq @ h1[b]  (4 partials per output, tree-combined)
  {
    int a = tid >> 2, p = tid & 3;
    const float* wr = wq + (size_t)a * HID_ + p * 256;
    const float* hr = h1 + (size_t)b * HID_ + p * 256;
    float s = 0.f;
    for (int k = 0; k < 256; ++k) s += wr[k] * hr[k];
    red[tid] = s;
  }
  __syncthreads();
  if (tid < ATT_)
    sq[tid] = red[tid * 4] + red[tid * 4 + 1] + red[tid * 4 + 2] + red[tid * 4 + 3];
  __syncthreads();

  const int s = tid;
  float feat[LOCF_];
#pragma unroll
  for (int f = 0; f < LOCF_; ++f) feat[f] = 0.f;
  for (int k = 0; k < LOCK_; ++k) {
    int p = s + k - (LOCK_ - 1) / 2;
    float va = 0.f, vc = 0.f;
    if (p >= 0 && p < S_) { va = sal[p]; vc = sac[p]; }
#pragma unroll
    for (int f = 0; f < LOCF_; ++f)
      feat[f] += va * slk[f * (2 * LOCK_) + k] + vc * slk[f * (2 * LOCK_) + LOCK_ + k];
  }
  float en = 0.f;
  const float* pmb = pm + ((size_t)b * S_ + s) * ATT_;
  for (int a = 0; a < ATT_; ++a) {
    float lf = 0.f;
#pragma unroll
    for (int f = 0; f < LOCF_; ++f) lf += swl[a * LOCF_ + f] * feat[f];
    en += sv[a] * tanhf(sq[a] + pmb[a] + lf);
  }
  // softmax over S
  red[tid] = en; __syncthreads();
  for (int o = 256; o > 0; o >>= 1) {
    if (tid < o) red[tid] = fmaxf(red[tid], red[tid + o]);
    __syncthreads();
  }
  float mx = red[0]; __syncthreads();
  float ex = expf(en - mx);
  red[tid] = ex; __syncthreads();
  for (int o = 256; o > 0; o >>= 1) {
    if (tid < o) red[tid] += red[tid + o];
    __syncthreads();
  }
  float aln = ex / red[0];
  __syncthreads();              // all conv reads of sal are done
  sal[s] = aln;
  al[b * S_ + s]  = aln;
  alc[b * S_ + s] = sac[s] + aln;
  alout[(size_t)b * T_ * S_ + s] = aln;     // alignments output
  __syncthreads();
  // ctx[d] = sum_s al[s] * enc[b,s,d]   (d = tid, coalesced over threads)
  const int d = tid;
  const float* eb = enc + (size_t)b * S_ * D_;
  float acc = 0.f;
  for (int j = 0; j < S_; ++j) acc += sal[j] * eb[(size_t)j * D_ + d];
  ctx[b * D_ + d]  = acc;
  ctxb[b * D_ + d] = (bf16)acc;
}

// ----------------------------- output heads --------------------------------
__global__ __launch_bounds__(128) void head_kernel(
    const float* __restrict__ c2, const float* __restrict__ ctx,
    const float* __restrict__ wm, const float* __restrict__ bm,
    const float* __restrict__ wsr, const float* __restrict__ bst,
    float* __restrict__ mel, float* __restrict__ stop, int t) {
  const int b = blockIdx.x, j = threadIdx.x;
  if (j > MEL_) return;
  const float* wrow = (j < MEL_) ? (wm + (size_t)j * (HID_ + D_)) : wsr;
  float acc = (j < MEL_) ? bm[j] : bst[0];
  const float* c2b = c2 + (size_t)b * HID_;
  const float* cxb = ctx + (size_t)b * D_;
  for (int k = 0; k < HID_; ++k) acc += wrow[k] * c2b[k];
  for (int k = 0; k < D_; ++k) acc += wrow[HID_ + k] * cxb[k];
  if (j < MEL_) mel[(size_t)b * MEL_ * T_ + (size_t)j * T_ + t] = acc;
  else          stop[(size_t)b * T_ + t] = acc;
}

// ------------------------------- launcher ----------------------------------
extern "C" void kernel_launch(void* const* d_in, const int* in_sizes, int n_in,
                              void* d_out, int out_size, void* d_ws,
                              size_t ws_size, hipStream_t stream) {
  const float* enc    = (const float*)d_in[0];
  const float* inputs = (const float*)d_in[1];
  const float* pw1 = (const float*)d_in[2];
  const float* pb1 = (const float*)d_in[3];
  const float* pw2 = (const float*)d_in[4];
  const float* pb2 = (const float*)d_in[5];
  const float* wih1 = (const float*)d_in[6];
  const float* whh1 = (const float*)d_in[7];
  const float* bih1 = (const float*)d_in[8];
  const float* bhh1 = (const float*)d_in[9];
  const float* wih2 = (const float*)d_in[10];
  const float* whh2 = (const float*)d_in[11];
  const float* bih2 = (const float*)d_in[12];
  const float* bhh2 = (const float*)d_in[13];
  const float* wq   = (const float*)d_in[14];
  const float* wv   = (const float*)d_in[15];
  const float* lk   = (const float*)d_in[16];
  const float* wloc = (const float*)d_in[17];
  const float* vat  = (const float*)d_in[18];
  const float* wm   = (const float*)d_in[19];
  const float* bm   = (const float*)d_in[20];
  const float* wsr  = (const float*)d_in[21];
  const float* bst  = (const float*)d_in[22];

  // workspace carve (all 256B-aligned; total ~74 MB -> L2-resident)
  char* base = (char*)d_ws;
  size_t off = 0;
  auto take = [&](size_t bytes) {
    char* p = base + off;
    off += (bytes + 255) & ~(size_t)255;
    return p;
  };
  bf16*  W1P   = (bf16*)take((size_t)NG_ * K1_ * 2);
  bf16*  W2P   = (bf16*)take((size_t)NG_ * K2_ * 2);
  bf16*  PRE16 = (bf16*)take((size_t)T_ * B_ * PRE_ * 2);
  float* PM    = (float*)take((size_t)B_ * S_ * ATT_ * 4);
  float* B1S   = (float*)take(NG_ * 4);
  float* B2S   = (float*)take(NG_ * 4);

  const size_t STATE_BYTES =
      4 * ((size_t)B_ * HID_ * 4) +   // h1 c1 h2 c2
      3 * ((size_t)B_ * S_ * 4) +     // ctx al alc  (S_==D_)
      4 * ((size_t)B_ * HID_ * 2) +   // h1b ping/pong, h2b ping/pong
      (size_t)B_ * D_ * 2;            // ctxb
  char* st = take(STATE_BYTES);
  char* st0 = st;
  float* h1  = (float*)st; st += (size_t)B_ * HID_ * 4;
  float* c1  = (float*)st; st += (size_t)B_ * HID_ * 4;
  float* h2  = (float*)st; st += (size_t)B_ * HID_ * 4;
  float* c2  = (float*)st; st += (size_t)B_ * HID_ * 4;
  float* ctx = (float*)st; st += (size_t)B_ * D_ * 4;
  float* al  = (float*)st; st += (size_t)B_ * S_ * 4;
  float* alc = (float*)st; st += (size_t)B_ * S_ * 4;
  bf16* h1ba = (bf16*)st;  st += (size_t)B_ * HID_ * 2;
  bf16* h1bb = (bf16*)st;  st += (size_t)B_ * HID_ * 2;
  bf16* h2ba = (bf16*)st;  st += (size_t)B_ * HID_ * 2;
  bf16* h2bb = (bf16*)st;  st += (size_t)B_ * HID_ * 2;
  bf16* ctxb = (bf16*)st;  st += (size_t)B_ * D_ * 2;

  // outputs (concatenated: mel [B,80,T], stop [B,T,1], align [B,T,S])
  float* mel_o  = (float*)d_out;
  float* stop_o = mel_o + (size_t)B_ * MEL_ * T_;
  float* alg_o  = stop_o + (size_t)B_ * T_;

  // --------- one-shot precompute (recomputed each call: deterministic) -----
  zero_kernel<<<512, 256, 0, stream>>>((float*)st0, STATE_BYTES / 4);
  bias_sum_kernel<<<16, 256, 0, stream>>>(bih1, bhh1, B1S, NG_);
  bias_sum_kernel<<<16, 256, 0, stream>>>(bih2, bhh2, B2S, NG_);
  pack_w_kernel<<<4096, 256, 0, stream>>>(wih1, whh1, W1P, PRE_ + D_, HID_);
  pack_w_kernel<<<4096, 256, 0, stream>>>(wih2, whh2, W2P, HID_ + D_, HID_);
  prenet_kernel<<<T_ * B_, 256, 0, stream>>>(inputs, pw1, pb1, pw2, pb2, PRE16);
  procmem_kernel<<<8192, 256, 0, stream>>>(enc, wv, PM);

  // ------------------------- sequential decode -----------------------------
  for (int t = 0; t < T_; ++t) {
    const bf16* h1r = (t & 1) ? h1bb : h1ba;
    bf16*       h1w = (t & 1) ? h1ba : h1bb;
    const bf16* h2r = (t & 1) ? h2bb : h2ba;
    bf16*       h2w = (t & 1) ? h2ba : h2bb;

    // LSTM1: A = [x_t(256) | ctx(512) | h1(1024)], K=1792
    lstm_kernel<<<64, 512, 0, stream>>>(
        W1P, B1S,
        PRE16 + (size_t)t * B_ * PRE_, PRE_, PRE_,
        ctxb, D_, D_,
        h1r, HID_, HID_,
        K1_, c1, h1, h1w);

    attn_kernel<<<64, 512, 0, stream>>>(
        h1, wq, lk, wloc, vat, PM, enc, al, alc, ctx, ctxb,
        alg_o + (size_t)t * S_);

    // LSTM2: A = [h1(1024) | ctx(512) | h2(1024)], K=2560
    lstm_kernel<<<64, 512, 0, stream>>>(
        W2P, B2S,
        h1w, HID_, HID_,
        ctxb, D_, D_,
        h2r, HID_, HID_,
        K2_, c2, h2, h2w);

    head_kernel<<<64, 128, 0, stream>>>(c2, ctx, wm, bm, wsr, bst,
                                        mel_o, stop_o, t);
  }
}
